// SmCoModel_75600014344328
// MI455X (gfx1250) — compile-verified, matrix operand order checked by reading, not run on hardware
//
#include <hip/hip_runtime.h>

// ---------------------------------------------------------------------------
// MoCo-isomap pipeline for MI455X (gfx1250, wave32, WMMA 16x16x32 f16->f32,
// TDM tensor_load_to_lds staging for the encoder weight tiles).
//
// Inputs (setup_inputs order, all fp32):
//  0 img_q [32,1,224,224]  1 img_k [32,1,224,224]
//  2 Wq [50176,128]        3 bq [128]
//  4 Wk [50176,128]        5 bk [128]
//  6 queue [128,512]
// Output: logits [32,513] fp32 ++ labels [32] (zeros)
// ---------------------------------------------------------------------------

#define BSZ    32
#define CHW    50176
#define FEAT   128
#define KQ     512
#define NV     513          // 1 query node + 512 queue nodes
#define NNODE  5
#define MMOM   0.999f
#define TINV   (1.0f / 0.07f)
#define NSLICE 98           // 50176 / 512
#define SLICEK 512
#define CHROWS 64           // K-rows per TDM chunk
#define NCH    (SLICEK / CHROWS)   // 8 chunks per slice

typedef __attribute__((ext_vector_type(16))) _Float16 v16h;
typedef __attribute__((ext_vector_type(8)))  float    v8f;
typedef __attribute__((ext_vector_type(4)))  unsigned v4u;
typedef __attribute__((ext_vector_type(8)))  int      v8i;
typedef __attribute__((ext_vector_type(4)))  int      v4i;

// Per-lane A/B fragment gather for V_WMMA_F32_16X16X32_F16 (ISA 7.12.2):
// half-wave hi: halves 0..7 hold K = base+hi*8+0..7, halves 8..15 hold
// K = base+hi*8+16+0..7 (two contiguous 16B runs -> b128 loads).
__device__ inline v16h frag_f16(const _Float16* rowp, int ka) {
  v16h f;
#pragma unroll
  for (int t = 0; t < 8; ++t) { f[t] = rowp[ka + t]; f[t + 8] = rowp[ka + 16 + t]; }
  return f;
}

// ---------------------------------------------------------------------------
// TDM: async-load a [rows x 16] f32 tile (row stride 128 floats) into LDS.
// D# layout per CDNA5 ISA ch.8 (group0: count/lds/global/type, group1: dims).
// ---------------------------------------------------------------------------
__device__ inline void tdm_load_tile(const float* gsrc, float* lds_dst, int rows) {
  const unsigned lds_off = (unsigned)(uintptr_t)lds_dst;   // low 32b = LDS offset
  const unsigned long long ga = (unsigned long long)(uintptr_t)gsrc;
  v4u g0;
  g0[0] = 1u;                                       // count=1 (valid user D#)
  g0[1] = lds_off;                                  // lds_addr [63:32]
  g0[2] = (unsigned)(ga & 0xffffffffu);             // global_addr [95:64]
  g0[3] = (unsigned)((ga >> 32) & 0x01ffffffu)      // global_addr [120:96]
        | 0x80000000u;                              // type=2 ("image") [127:126]
  v8i g1;
  g1[0] = 0x20000;                // workgroup_mask=0, data_size=2 (4 bytes)
  g1[1] = 16 << 16;               // tensor_dim0 = 16  -> bits [63:48]
  g1[2] = 512 << 16;              // tensor_dim1 = 512 -> bits [95:80]
  g1[3] = 16 << 16;               // tile_dim0 = 16    -> bits [127:112]
  g1[4] = rows & 0xffff;          // tile_dim1 = rows; tile_dim2 = 0
  g1[5] = 128;                    // tensor_dim0_stride = 128 elements
  g1[6] = 0;                      // stride hi / dim1_stride lo (2D: unused)
  g1[7] = 0;
  v4i z4 = {0, 0, 0, 0};
#if __clang_major__ >= 23
  v8i z8 = {0, 0, 0, 0, 0, 0, 0, 0};
  __builtin_amdgcn_tensor_load_to_lds(g0, g1, z4, z4, z8, 0);
#else
  __builtin_amdgcn_tensor_load_to_lds(g0, g1, z4, z4, 0);
#endif
}

// ---------------------------------------------------------------------------
// K-split encoder GEMM: out[32,128] = img[32,50176] @ W (+ momentum combine).
// One wave per (16x16 tile, 512-wide K slice). W tiles staged to LDS by the
// Tensor Data Mover, double-buffered on TENSORcnt; A streamed as b128 loads.
// Partials reduced deterministically in a second kernel.
// ---------------------------------------------------------------------------
template <int IS_K>
__global__ void __launch_bounds__(32)
encode_wmma(const float* __restrict__ img,
            const float* __restrict__ Wq, const float* __restrict__ Wk,
            float* __restrict__ part) {
  __shared__ float wq_tile[2][CHROWS * 16];
  __shared__ float wk_tile[IS_K ? 2 : 1][CHROWS * 16];
  const int lane = threadIdx.x;
  const int nx = blockIdx.x;            // N tile 0..7
  const int my = blockIdx.y;            // M tile 0..1
  const int slice = blockIdx.z;         // K slice 0..97
  const int r  = lane & 15;
  const int hi = lane >> 4;
  const int n0 = nx * 16;
  const float* arow = img + (size_t)(my * 16 + r) * CHW;
  const int k0 = slice * SLICEK;
  const int NLOAD = IS_K ? 2 : 1;       // TDM ops in flight per chunk

  // prologue: DMA chunk 0
  tdm_load_tile(Wq + (size_t)k0 * FEAT + n0, &wq_tile[0][0], CHROWS);
  if (IS_K) tdm_load_tile(Wk + (size_t)k0 * FEAT + n0, &wk_tile[0][0], CHROWS);

  v8f c = {};
#pragma unroll
  for (int chk = 0; chk < NCH; ++chk) {
    const int cb = chk & 1;
    if (chk + 1 < NCH) {                // DMA next chunk, then wait for current
      const size_t goff = (size_t)(k0 + (chk + 1) * CHROWS) * FEAT + n0;
      tdm_load_tile(Wq + goff, &wq_tile[cb ^ 1][0], CHROWS);
      if (IS_K) tdm_load_tile(Wk + goff, &wk_tile[IS_K ? (cb ^ 1) : 0][0], CHROWS);
      __builtin_amdgcn_s_wait_tensorcnt((short)(IS_K ? 2 : 1));
    } else {
      __builtin_amdgcn_s_wait_tensorcnt((short)0);
    }
    const float* wq = &wq_tile[cb][0];
    const float* wk = &wk_tile[IS_K ? cb : 0][0];
#pragma unroll
    for (int st = 0; st < CHROWS / 32; ++st) {     // 2 WMMA k-steps per chunk
      const int kk = k0 + chk * CHROWS + st * 32;  // global K base
      const int ka = kk + hi * 8;
      __builtin_prefetch(arow + kk + 64, 0, 1);    // global_prefetch_b8
      // A fragment: 4x b128 loads + f16 convert
      const float4 x0 = *(const float4*)(arow + ka);
      const float4 x1 = *(const float4*)(arow + ka + 4);
      const float4 x2 = *(const float4*)(arow + ka + 16);
      const float4 x3 = *(const float4*)(arow + ka + 20);
      v16h a, b;
      a[0] = (_Float16)x0.x; a[1] = (_Float16)x0.y;
      a[2] = (_Float16)x0.z; a[3] = (_Float16)x0.w;
      a[4] = (_Float16)x1.x; a[5] = (_Float16)x1.y;
      a[6] = (_Float16)x1.z; a[7] = (_Float16)x1.w;
      a[8]  = (_Float16)x2.x; a[9]  = (_Float16)x2.y;
      a[10] = (_Float16)x2.z; a[11] = (_Float16)x2.w;
      a[12] = (_Float16)x3.x; a[13] = (_Float16)x3.y;
      a[14] = (_Float16)x3.z; a[15] = (_Float16)x3.w;
      // B fragment from LDS chunk (rows of 16 floats; conflict-light)
      const int lk = st * 32 + hi * 8;             // chunk-local K
#pragma unroll
      for (int t = 0; t < 8; ++t) {
        float w1 = wq[(lk + t) * 16 + r];
        float w2 = wq[(lk + 16 + t) * 16 + r];
        if (IS_K) {                                // key enc: M*Wk + (1-M)*Wq
          w1 = MMOM * wk[(lk + t) * 16 + r]      + (1.0f - MMOM) * w1;
          w2 = MMOM * wk[(lk + 16 + t) * 16 + r] + (1.0f - MMOM) * w2;
        }
        b[t] = (_Float16)w1; b[t + 8] = (_Float16)w2;
      }
      c = __builtin_amdgcn_wmma_f32_16x16x32_f16(false, a, false, b,
                                                 (short)0, c, false, false);
    }
  }
  const int tl = (IS_K * 2 + my) * 8 + nx;        // 0..31 tile id
#pragma unroll
  for (int j = 0; j < 8; ++j) {
    const int local = (j + 8 * hi) * 16 + (lane & 15);
    part[((size_t)(tl * 256 + local)) * NSLICE + slice] = c[j];
  }
}

// Deterministic fixed-order reduction over the 98 K-slice partials.
__global__ void __launch_bounds__(256)
encode_reduce(const float* __restrict__ part,
              float* __restrict__ q_raw, float* __restrict__ k_raw) {
  const int tid = blockIdx.x * blockDim.x + threadIdx.x;   // 0..8191
  const int tl = tid >> 8, local = tid & 255;
  float s = 0.f;
  for (int sl = 0; sl < NSLICE; ++sl) s += part[(size_t)tid * NSLICE + sl];
  const int is_k = tl >> 4;
  const int row = ((tl >> 3) & 1) * 16 + (local >> 4);
  const int col = (tl & 7) * 16 + (local & 15);
  (is_k ? k_raw : q_raw)[row * FEAT + col] = s;
}

// ---------------------------------------------------------------------------
// Bias + L2-normalize q/k; emit q in f16 + squared norms.
// ---------------------------------------------------------------------------
__global__ void __launch_bounds__(32)
bias_normalize(const float* __restrict__ q_raw, const float* __restrict__ k_raw,
               const float* __restrict__ bq, const float* __restrict__ bk,
               float* __restrict__ qn, float* __restrict__ kn,
               _Float16* __restrict__ qf16, float* __restrict__ sq_q) {
  const int b = blockIdx.x, z = blockIdx.y, lane = threadIdx.x;
  const float* in = z ? k_raw : q_raw;
  float v[4], s = 0.f;
#pragma unroll
  for (int t = 0; t < 4; ++t) {
    const int c = lane * 4 + t;
    const float bias = z ? (MMOM * bk[c] + (1.0f - MMOM) * bq[c]) : bq[c];
    v[t] = in[b * FEAT + c] + bias;
    s += v[t] * v[t];
  }
  for (int off = 16; off; off >>= 1) s += __shfl_xor(s, off, 32);
  const float denom = fmaxf(sqrtf(s), 1e-12f);
  float* outp = z ? kn : qn;
#pragma unroll
  for (int t = 0; t < 4; ++t) {
    const int c = lane * 4 + t;
    const float u = v[t] / denom;
    outp[b * FEAT + c] = u;
    if (!z) qf16[b * FEAT + c] = (_Float16)u;
  }
  if (!z && lane == 0) sq_q[b] = s / (denom * denom);
}

// queue [128,512] -> queueT f16 [512,128] (+ per-column squared norm)
__global__ void __launch_bounds__(32)
queue_prep(const float* __restrict__ queue, _Float16* __restrict__ queueT,
           float* __restrict__ sq_queue) {
  const int r = blockIdx.x, lane = threadIdx.x;
  float s = 0.f;
  for (int c = lane; c < FEAT; c += 32) {
    const float v = queue[(size_t)c * KQ + r];
    queueT[(size_t)r * FEAT + c] = (_Float16)v;
    s += v * v;
  }
  for (int off = 16; off; off >>= 1) s += __shfl_xor(s, off, 32);
  if (lane == 0) sq_queue[r] = s;
}

// ---------------------------------------------------------------------------
// Batched gram + euclidean distance: d[b,i,j] = sqrt(|ai|^2+|aj|^2-2 ai.aj)
// One wave per 16x16 (i,j) tile; K=128 -> 4 WMMA steps.
// ---------------------------------------------------------------------------
__global__ void __launch_bounds__(32)
gram_dist(const _Float16* __restrict__ qf16, const _Float16* __restrict__ queueT,
          const float* __restrict__ sq_q, const float* __restrict__ sq_queue,
          float* __restrict__ dbuf) {
  const int lane = threadIdx.x;
  const int i0 = blockIdx.x * 16, j0 = blockIdx.y * 16, b = blockIdx.z;
  const int r = lane & 15, hi = lane >> 4;
  const int ia = i0 + r, jb = j0 + r;
  const _Float16* arow = (ia == 0) ? (qf16 + (size_t)b * FEAT)
                       : (ia <= KQ ? queueT + (size_t)(ia - 1) * FEAT : nullptr);
  const _Float16* brow = (jb == 0) ? (qf16 + (size_t)b * FEAT)
                       : (jb <= KQ ? queueT + (size_t)(jb - 1) * FEAT : nullptr);
  v8f c = {};
  for (int kk = 0; kk < FEAT; kk += 32) {
    const int ka = kk + hi * 8;
    v16h a = {}, bm = {};
    if (arow) a  = frag_f16(arow, ka);
    if (brow) bm = frag_f16(brow, ka);
    c = __builtin_amdgcn_wmma_f32_16x16x32_f16(false, a, false, bm,
                                               (short)0, c, false, false);
  }
#pragma unroll
  for (int j = 0; j < 8; ++j) {
    const int i  = i0 + j + 8 * hi;
    const int jj = j0 + (lane & 15);
    if (i < NV && jj < NV) {
      const float sqi = (i == 0)  ? sq_q[b] : sq_queue[i - 1];
      const float sqj = (jj == 0) ? sq_q[b] : sq_queue[jj - 1];
      const float d2 = sqi + sqj - 2.0f * c[j];
      dbuf[((size_t)b * NV + i) * NV + jj] = sqrtf(fmaxf(d2, 0.0f));
    }
  }
}

// ---------------------------------------------------------------------------
// Per-row top-5 (== ranks >= n-5 from stable argsort-of-argsort: (value,index)
// lexicographic, larger index wins ties). One wave per (row, batch).
// ---------------------------------------------------------------------------
__global__ void __launch_bounds__(32)
top5_edges(const float* __restrict__ dbuf,
           int* __restrict__ eidx, float* __restrict__ ew) {
  const int i = blockIdx.x, b = blockIdx.y, lane = threadIdx.x;
  const float* row = dbuf + ((size_t)b * NV + i) * NV;
  int chosen[NNODE];
#pragma unroll
  for (int s = 0; s < NNODE; ++s) chosen[s] = -1;
  for (int s = 0; s < NNODE; ++s) {
    float bv = -1.0f; int bj = -1;
    for (int j = lane; j < NV; j += 32) {
      bool taken = false;
#pragma unroll
      for (int t = 0; t < NNODE; ++t) taken |= (chosen[t] == j);
      if (taken) continue;
      const float v = row[j];
      if (v > bv || (v == bv && j > bj)) { bv = v; bj = j; }
    }
    for (int off = 16; off; off >>= 1) {      // butterfly: all lanes converge
      const float ov = __shfl_xor(bv, off, 32);
      const int   oj = __shfl_xor(bj, off, 32);
      if (ov > bv || (ov == bv && oj > bj)) { bv = ov; bj = oj; }
    }
    chosen[s] = bj;
    if (lane == 0) {
      // edge rule: weight>0, or ==0 self-loop (no-op for SSSP -> drop)
      const bool valid = (bj >= 0) && (bv > 0.0f);
      eidx[((size_t)b * NV + i) * NNODE + s] = valid ? bj : -1;
      ew  [((size_t)b * NV + i) * NNODE + s] = valid ? bv : 0.0f;
    }
  }
}

// ---------------------------------------------------------------------------
// 512 Jacobi Bellman-Ford relaxations from node 0 (== reference fori_loop).
// Sparse: 5 edges/row cached in LDS; uint atomicMin is order-correct for
// nonnegative floats (incl. +inf). Double-buffered for Jacobi semantics.
// ---------------------------------------------------------------------------
__global__ void __launch_bounds__(256)
bellman_ford(const int* __restrict__ eidx, const float* __restrict__ ew,
             float* __restrict__ dist_out) {
  __shared__ unsigned dA[NV], dB[NV];
  __shared__ int   se[NV * NNODE];
  __shared__ float swt[NV * NNODE];
  const int b = blockIdx.x, t = threadIdx.x;
  const unsigned INFU = 0x7f800000u;
  for (int e = t; e < NV * NNODE; e += blockDim.x) {
    se[e]  = eidx[(size_t)b * NV * NNODE + e];
    swt[e] = ew  [(size_t)b * NV * NNODE + e];
  }
  for (int i = t; i < NV; i += blockDim.x) dA[i] = (i == 0) ? 0u : INFU;
  __syncthreads();
  unsigned* cur = dA; unsigned* nxt = dB;
  for (int it = 0; it < NV - 1; ++it) {
    for (int i = t; i < NV; i += blockDim.x) nxt[i] = cur[i];
    __syncthreads();
    for (int i = t; i < NV; i += blockDim.x) {
      const unsigned du = cur[i];
      if (du < INFU) {
        const float df = __uint_as_float(du);
#pragma unroll
        for (int s = 0; s < NNODE; ++s) {
          const int j = se[i * NNODE + s];
          if (j >= 0)
            atomicMin(&nxt[j], __float_as_uint(df + swt[i * NNODE + s]));
        }
      }
    }
    __syncthreads();
    unsigned* tmp = cur; cur = nxt; nxt = tmp;
  }
  for (int i = t; i < KQ; i += blockDim.x)
    dist_out[(size_t)b * KQ + i] = __uint_as_float(cur[i + 1]);
}

// Global max over finite distances (+1), single block.
__global__ void __launch_bounds__(256)
reduce_max_finite(const float* __restrict__ dist, float* __restrict__ mx1) {
  __shared__ float sm[256];
  const int t = threadIdx.x;
  float m = 0.f;
  for (int i = t; i < BSZ * KQ; i += 256) {
    const float v = dist[i];
    if (v < __builtin_inff()) m = fmaxf(m, v);
  }
  sm[t] = m; __syncthreads();
  for (int off = 128; off; off >>= 1) {
    if (t < off) sm[t] = fmaxf(sm[t], sm[t + off]);
    __syncthreads();
  }
  if (t == 0) mx1[0] = sm[0] + 1.0f;
}

__global__ void __launch_bounds__(256)
make_weight(const float* __restrict__ dist, const float* __restrict__ mx1,
            float* __restrict__ weight) {
  const int i = blockIdx.x * blockDim.x + threadIdx.x;
  if (i < BSZ * KQ) {
    float v = dist[i];
    if (!(v < __builtin_inff())) v = mx1[0];
    weight[i] = 1.0f / (1.0f + v);
  }
}

// ---------------------------------------------------------------------------
// l_neg = (q @ queue) * weight / T via WMMA; one wave per 16x16 tile.
// ---------------------------------------------------------------------------
__global__ void __launch_bounds__(32)
logits_neg(const _Float16* __restrict__ qf16, const _Float16* __restrict__ queueT,
           const float* __restrict__ weight, float* __restrict__ out) {
  const int lane = threadIdx.x;
  const int n0 = blockIdx.x * 16, m0 = blockIdx.y * 16;
  const int r = lane & 15, hi = lane >> 4;
  const _Float16* arow = qf16 + (size_t)(m0 + r) * FEAT;
  const _Float16* brow = queueT + (size_t)(n0 + r) * FEAT;
  v8f c = {};
  for (int kk = 0; kk < FEAT; kk += 32) {
    const int ka = kk + hi * 8;
    v16h a = frag_f16(arow, ka), bm = frag_f16(brow, ka);
    c = __builtin_amdgcn_wmma_f32_16x16x32_f16(false, a, false, bm,
                                               (short)0, c, false, false);
  }
#pragma unroll
  for (int j = 0; j < 8; ++j) {
    const int row = m0 + j + 8 * hi;
    const int col = n0 + (lane & 15);
    out[(size_t)row * NV + 1 + col] = c[j] * weight[row * KQ + col] * TINV;
  }
}

// l_pos = sum(q*k)/T into out[b*513]; labels (zeros) appended after logits.
__global__ void __launch_bounds__(32)
lpos_labels(const float* __restrict__ qn, const float* __restrict__ kn,
            float* __restrict__ out) {
  const int b = blockIdx.x, lane = threadIdx.x;
  float s = 0.f;
#pragma unroll
  for (int t = 0; t < 4; ++t) {
    const int c = lane * 4 + t;
    s += qn[b * FEAT + c] * kn[b * FEAT + c];
  }
  for (int off = 16; off; off >>= 1) s += __shfl_xor(s, off, 32);
  if (lane == 0) {
    out[(size_t)b * NV] = s * TINV;
    out[(size_t)BSZ * NV + b] = 0.0f;   // int32 zero bit-pattern == 0.0f
  }
}

// ---------------------------------------------------------------------------
extern "C" void kernel_launch(void* const* d_in, const int* in_sizes, int n_in,
                              void* d_out, int out_size, void* d_ws, size_t ws_size,
                              hipStream_t stream) {
  const float* img_q = (const float*)d_in[0];
  const float* img_k = (const float*)d_in[1];
  const float* Wq    = (const float*)d_in[2];
  const float* bq    = (const float*)d_in[3];
  const float* Wk    = (const float*)d_in[4];
  const float* bk    = (const float*)d_in[5];
  const float* queue = (const float*)d_in[6];
  float* out = (float*)d_out;

  // bump allocator over workspace (all buffers fully written before read)
  size_t off = 0;
  auto alloc = [&](size_t bytes) {
    void* p = (char*)d_ws + off;
    off = (off + bytes + 255) & ~(size_t)255;
    return p;
  };
  float*    part     = (float*)   alloc((size_t)32 * 256 * NSLICE * 4);   // 3.2 MB
  float*    q_raw    = (float*)   alloc(BSZ * FEAT * 4);
  float*    k_raw    = (float*)   alloc(BSZ * FEAT * 4);
  float*    qn       = (float*)   alloc(BSZ * FEAT * 4);
  float*    kn       = (float*)   alloc(BSZ * FEAT * 4);
  _Float16* qf16     = (_Float16*)alloc(BSZ * FEAT * 2);
  float*    sq_q     = (float*)   alloc(BSZ * 4);
  _Float16* queueT   = (_Float16*)alloc((size_t)KQ * FEAT * 2);
  float*    sq_queue = (float*)   alloc(KQ * 4);
  float*    dbuf     = (float*)   alloc((size_t)BSZ * NV * NV * 4);       // 33.7 MB
  int*      eidx     = (int*)     alloc((size_t)BSZ * NV * NNODE * 4);
  float*    ew       = (float*)   alloc((size_t)BSZ * NV * NNODE * 4);
  float*    dist     = (float*)   alloc((size_t)BSZ * KQ * 4);
  float*    mx1      = (float*)   alloc(256);
  float*    weight   = (float*)   alloc((size_t)BSZ * KQ * 4);
  (void)ws_size; (void)in_sizes; (void)n_in; (void)out_size;

  // 1) encoders (K-split WMMA GEMM, TDM-staged weights, deterministic reduce)
  encode_wmma<0><<<dim3(8, 2, NSLICE), 32, 0, stream>>>(img_q, Wq, Wk, part);
  encode_wmma<1><<<dim3(8, 2, NSLICE), 32, 0, stream>>>(img_k, Wq, Wk, part);
  encode_reduce<<<32, 256, 0, stream>>>(part, q_raw, k_raw);
  bias_normalize<<<dim3(BSZ, 2), 32, 0, stream>>>(q_raw, k_raw, bq, bk,
                                                  qn, kn, qf16, sq_q);
  // 2) isomap weights
  queue_prep<<<KQ, 32, 0, stream>>>(queue, queueT, sq_queue);
  gram_dist<<<dim3(33, 33, BSZ), 32, 0, stream>>>(qf16, queueT, sq_q, sq_queue, dbuf);
  top5_edges<<<dim3(NV, BSZ), 32, 0, stream>>>(dbuf, eidx, ew);
  bellman_ford<<<BSZ, 256, 0, stream>>>(eidx, ew, dist);
  reduce_max_finite<<<1, 256, 0, stream>>>(dist, mx1);
  make_weight<<<(BSZ * KQ + 255) / 256, 256, 0, stream>>>(dist, mx1, weight);
  // 3) logits
  logits_neg<<<dim3(KQ / 16, BSZ / 16), 32, 0, stream>>>(qf16, queueT, weight, out);
  lpos_labels<<<BSZ, 32, 0, stream>>>(qn, kn, out);
}
